// ModelLSTM_3384434229934
// MI455X (gfx1250) — matrix operand order repaired and checked by reading, not compiled, not run on hardware
//
#include <hip/hip_runtime.h>

#define Tlen 512
#define NT   28        // 448/16 gate tiles
#define KP   128       // padded K (H=100 -> 128)

typedef unsigned short u16;
typedef __attribute__((ext_vector_type(16))) __bf16 v16bf;
typedef __attribute__((ext_vector_type(8)))  float  v8f;

union ABU { v16bf v; uint4 q[2]; };

__device__ __forceinline__ float bf2f(u16 u) {
    union { float f; unsigned int i; } x; x.i = ((unsigned int)u) << 16; return x.f;
}
__device__ __forceinline__ u16 f2bf(float f) {
    union { float f; unsigned int i; } x; x.f = f;
    unsigned int r = x.i + 0x7FFFu + ((x.i >> 16) & 1u);
    return (u16)(r >> 16);
}
__device__ __forceinline__ float sigm(float x) { return 1.0f / (1.0f + __expf(-x)); }

// ---------------------------------------------------------------------------
// Generic WMMA GEMM:  C[m,n] (+)= sum_k A[m+rowOff, k] * Bt[n, k]
// A, Bt bf16 row-major (inner dim = K, 16B-aligned rows). One wave per 16x16
// output tile; K consumed in chunks of 32 via v_wmma_f32_16x16x32_bf16.
// Row-shift validity (conv taps) is wave-uniform (rowOff is a multiple of the
// 256-row time block), so it is resolved with a scalar branch that skips the
// whole K loop instead of per-lane exec masking.
// Output: either f32 row-major (optional accumulate) or bf16 wave-tiled
// (layout: tile*256 + r*32 + lane), matching what lstm_scan reads back.
// ---------------------------------------------------------------------------
__global__ void gemm_tn(const u16* __restrict__ A, int lda,
                        const u16* __restrict__ Bt, int ldb,
                        int Mtiles, int Ntiles, int Kchunks,
                        int Mrows, int rowOff,
                        float* __restrict__ Cf32, int ldc, int accumulate,
                        u16* __restrict__ Ctiled)
{
    int wave = blockIdx.x * (blockDim.x >> 5) + (threadIdx.x >> 5);
    int lane = threadIdx.x & 31;
    if (wave >= Mtiles * Ntiles) return;
    int mt = __builtin_amdgcn_readfirstlane(wave / Ntiles);
    int nt = __builtin_amdgcn_readfirstlane(wave % Ntiles);

    // wave-uniform tile validity (16-row tiles never straddle the range)
    int rowBase = mt * 16 + rowOff;
    bool vr = (rowBase >= 0) && (rowBase + 15 < Mrows);

    int rowA = rowBase + (lane & 15);
    int nB   = nt * 16 + (lane & 15);
    int hi   = lane >> 4;

    v8f c;
    if (Ctiled || !accumulate) {
        #pragma unroll
        for (int r = 0; r < 8; r++) c[r] = 0.0f;
    } else {
        int row0 = mt * 16 + hi * 8;
        int col  = nt * 16 + (lane & 15);
        #pragma unroll
        for (int r = 0; r < 8; r++) c[r] = Cf32[(size_t)(row0 + r) * ldc + col];
    }

    if (vr) {                         // scalar branch: zero A-tile adds nothing
        const u16* arow = A + (size_t)rowA * lda + hi * 8;   // e0..7 @ +0, e8..15 @ +16
        const u16* brow = Bt + (size_t)nB * ldb + hi * 8;    // e0..7 @ +0, e8..15 @ +16
        #pragma unroll 4
        for (int kc = 0; kc < Kchunks; kc++) {
            ABU ra, rb;
            ra.q[0] = *(const uint4*)(arow);
            ra.q[1] = *(const uint4*)(arow + 16);
            rb.q[0] = *(const uint4*)(brow);
            rb.q[1] = *(const uint4*)(brow + 16);
            arow += 32; brow += 32;
            c = __builtin_amdgcn_wmma_f32_16x16x32_bf16(false, ra.v, false, rb.v,
                                                        (short)0, c, false, false);
        }
    }

    if (Ctiled) {
        u16* o = Ctiled + (size_t)wave * 256 + lane;
        #pragma unroll
        for (int r = 0; r < 8; r++) o[r * 32] = f2bf(c[r]);
    } else {
        int row0 = mt * 16 + hi * 8;
        int col  = nt * 16 + (lane & 15);
        #pragma unroll
        for (int r = 0; r < 8; r++) Cf32[(size_t)(row0 + r) * ldc + col] = c[r];
    }
}

// ---------------------------------------------------------------------------
// Persistent LSTM time scan. grid=16 blocks x 32 threads: one wave owns 16
// batch rows (independent recurrence). h lives in a private LDS slab (bf16,
// 16x128); c lives in VGPRs. Per step: 4 ds_load_b128 A-tiles, 28x4 WMMA
// against Whh (L2-resident), fused sigmoid/tanh epilogue, h written back to
// LDS + global ys. No barriers needed: per-wave LDS ops are in-order.
// ---------------------------------------------------------------------------
__global__ void __launch_bounds__(32)
lstm_scan(const u16* __restrict__ G,        // [T*16][28][256] bf16 wave-tiled
          const u16* __restrict__ WhhBt,    // [448][128] bf16
          const float* __restrict__ bias,   // [448] (bih+bhh, gate-padded)
          const u16* __restrict__ h_init,   // [256][128] bf16
          const float* __restrict__ c_init, // wave-tiled or null
          int c_valid,
          u16* __restrict__ ys,             // [T*256][128] bf16 (cols 112.. stay 0)
          u16* __restrict__ h_fin,          // [256][128] bf16
          float* __restrict__ c_out, int T)
{
    __shared__ u16 hbuf[16 * 128];
    const int lane = threadIdx.x;
    const int mt   = blockIdx.x;            // batch tile 0..15
    const int rit  = lane & 15;
    const int hi   = lane >> 4;

    for (int i = lane; i < 256; i += 32)    // 256 uint4 = 16x128 bf16
        ((uint4*)hbuf)[i] = ((const uint4*)(h_init + (size_t)mt * 2048))[i];

    float c_reg[7][8];
    #pragma unroll
    for (int jt = 0; jt < 7; jt++)
        #pragma unroll
        for (int r = 0; r < 8; r++)
            c_reg[jt][r] = c_valid ? c_init[((mt * 7 + jt) * 8 + r) * 32 + lane] : 0.0f;

    float bi[7], bff[7], bg[7], bo[7];
    #pragma unroll
    for (int jt = 0; jt < 7; jt++) {
        int col = jt * 16 + rit;
        bi[jt]  = bias[col];
        bff[jt] = bias[112 + col];
        bg[jt]  = bias[224 + col];
        bo[jt]  = bias[336 + col];
    }

    for (int t = 0; t < T; t++) {
        v16bf a[4];
        #pragma unroll
        for (int kc = 0; kc < 4; kc++) {
            ABU r;
            const u16* hp = hbuf + rit * 128 + kc * 32 + hi * 8;
            r.q[0] = *(const uint4*)(hp);
            r.q[1] = *(const uint4*)(hp + 16);
            a[kc] = r.v;
        }
        const u16* Gt = G + ((size_t)(t * 16 + mt) * NT) * 256;

        #pragma unroll
        for (int jt = 0; jt < 7; jt++) {
            v8f acc[4];
            #pragma unroll
            for (int g = 0; g < 4; g++) {
                int nt = g * 7 + jt;
                const u16* gp = Gt + nt * 256 + lane;
                v8f c;
                #pragma unroll
                for (int r = 0; r < 8; r++) c[r] = bf2f(gp[r * 32]);
                #pragma unroll
                for (int kc = 0; kc < 4; kc++) {
                    ABU rb;
                    const u16* bp = WhhBt + (size_t)(nt * 16 + rit) * KP + kc * 32 + hi * 8;
                    rb.q[0] = *(const uint4*)(bp);
                    rb.q[1] = *(const uint4*)(bp + 16);
                    c = __builtin_amdgcn_wmma_f32_16x16x32_bf16(false, a[kc], false, rb.v,
                                                                (short)0, c, false, false);
                }
                acc[g] = c;
            }
            int col = jt * 16 + rit;
            bool vc = col < 100;
            #pragma unroll
            for (int r = 0; r < 8; r++) {
                float iv = sigm(acc[0][r] + bi[jt]);
                float fv = sigm(acc[1][r] + bff[jt]);
                float gv = tanhf(acc[2][r] + bg[jt]);
                float ov = sigm(acc[3][r] + bo[jt]);
                float cn = fv * c_reg[jt][r] + iv * gv;
                float hv = ov * tanhf(cn);
                if (!vc) { cn = 0.0f; hv = 0.0f; }
                c_reg[jt][r] = cn;
                u16 hb = f2bf(hv);
                int row = r + hi * 8;
                hbuf[row * 128 + col] = hb;
                ys[((size_t)(t * 256) + mt * 16 + row) * 128 + col] = hb;
            }
        }
    }

    #pragma unroll
    for (int jt = 0; jt < 7; jt++)
        #pragma unroll
        for (int r = 0; r < 8; r++)
            c_out[((mt * 7 + jt) * 8 + r) * 32 + lane] = c_reg[jt][r];
    for (int i = lane; i < 256; i += 32)
        ((uint4*)(h_fin + (size_t)mt * 2048))[i] = ((const uint4*)hbuf)[i];
}

// ------------------------------ pack / epilogue kernels ---------------------
__global__ void pack_x(const float* __restrict__ x, u16* __restrict__ A0)
{   // x [B=256,T,4] -> A0 [T*256][32] bf16 (time-major, K padded 4->32)
    int idx = blockIdx.x * blockDim.x + threadIdx.x;
    if (idx >= Tlen * 256 * 32) return;
    int k = idx & 31, m = idx >> 5;
    int t = m >> 8, b = m & 255;
    float v = (k < 4) ? x[((size_t)b * Tlen + t) * 4 + k] : 0.0f;
    A0[idx] = f2bf(v);
}

__global__ void pack_gate_weight(const float* __restrict__ src, u16* __restrict__ dst,
                                 int Ksrc, int Kp)
{   // src [400][Ksrc] -> dst [448][Kp] bf16, gates padded 100->112, K padded
    int idx = blockIdx.x * blockDim.x + threadIdx.x;
    if (idx >= 448 * Kp) return;
    int n = idx / Kp, k = idx % Kp;
    int g = n / 112, j = n % 112;
    float v = (j < 100 && k < Ksrc) ? src[(size_t)(g * 100 + j) * Ksrc + k] : 0.0f;
    dst[idx] = f2bf(v);
}

__global__ void pack_bias(const float* __restrict__ bih, const float* __restrict__ bhh,
                          float* __restrict__ dst)
{
    int n = blockIdx.x * blockDim.x + threadIdx.x;
    if (n >= 448) return;
    int g = n / 112, j = n % 112;
    dst[n] = (j < 100) ? bih[g * 100 + j] + bhh[g * 100 + j] : 0.0f;
}

__global__ void pack_conv(const float* __restrict__ w, u16* __restrict__ dst, int taps)
{   // w [100][100][taps] -> dst [taps][112][128] bf16
    int idx = blockIdx.x * blockDim.x + threadIdx.x;
    if (idx >= taps * 112 * 128) return;
    int tap = idx / (112 * 128);
    int r = idx % (112 * 128);
    int co = r / 128, ci = r % 128;
    float v = (co < 100 && ci < 100) ? w[((size_t)co * 100 + ci) * taps + tap] : 0.0f;
    dst[idx] = f2bf(v);
}

__global__ void relu_bf16(const u16* __restrict__ in, u16* __restrict__ out, int n)
{
    int i = blockIdx.x * blockDim.x + threadIdx.x;
    if (i < n) { u16 v = in[i]; out[i] = (v & 0x8000u) ? (u16)0 : v; }
}

__global__ void conv_finish(const float* __restrict__ C, const float* __restrict__ bias,
                            u16* __restrict__ out, int M)
{   // C [M][112] f32 + bias -> out [M][128] bf16 (padded zeros)
    int idx = blockIdx.x * blockDim.x + threadIdx.x;
    if (idx >= M * 128) return;
    int col = idx & 127, m = idx >> 7;
    float v = (col < 100) ? C[(size_t)m * 112 + col] + bias[col] : 0.0f;
    out[idx] = f2bf(v);
}

__global__ void linear_out(const float* __restrict__ C, const float* __restrict__ cbias,
                           const float* __restrict__ lw, const float* __restrict__ lb,
                           float* __restrict__ out)
{   // out[b,t,j] = sum_k (C[t*256+b,k]+cbias[k]) * lw[j,k] + lb[j]
    int idx = blockIdx.x * blockDim.x + threadIdx.x;
    if (idx >= Tlen * 256 * 4) return;
    int j = idx & 3, m = idx >> 2;
    int t = m >> 8, b = m & 255;
    const float* row = C + (size_t)m * 112;
    float s = lb[j];
    for (int k = 0; k < 100; k++) s += (row[k] + cbias[k]) * lw[j * 100 + k];
    out[((size_t)b * Tlen + t) * 4 + j] = s;
}

// ---------------------------------------------------------------------------
extern "C" void kernel_launch(void* const* d_in, const int* in_sizes, int n_in,
                              void* d_out, int out_size, void* d_ws, size_t ws_size,
                              hipStream_t stream)
{
    const float* x     = (const float*)d_in[0];
    const float* wih0  = (const float*)d_in[1];
    const float* wih1  = (const float*)d_in[2];   // [9][400][100]
    const float* whh1  = (const float*)d_in[3];   // [10][400][100]
    const float* bih1  = (const float*)d_in[4];
    const float* bhh1  = (const float*)d_in[5];
    const float* wih2  = (const float*)d_in[6];
    const float* whh2  = (const float*)d_in[7];
    const float* bih2  = (const float*)d_in[8];
    const float* bhh2  = (const float*)d_in[9];
    const float* c1w   = (const float*)d_in[10];
    const float* c1b   = (const float*)d_in[11];
    const float* c2w   = (const float*)d_in[12];
    const float* c2b   = (const float*)d_in[13];
    const float* lw    = (const float*)d_in[14];
    const float* lb    = (const float*)d_in[15];
    float* out = (float*)d_out;

    char* p = (char*)d_ws;
    auto alloc = [&](size_t bytes) -> char* {
        char* r = p; p += (bytes + 255) & ~(size_t)255; return r;
    };
    u16*   zerosH = (u16*)  alloc(256 * 128 * 2);
    u16*   hfin   = (u16*)  alloc(10 * 256 * 128 * 2);
    float* cbuf   = (float*)alloc(10 * 256 * 112 * 4);
    float* bias1  = (float*)alloc(10 * 448 * 4);
    float* bias2  = (float*)alloc(10 * 448 * 4);
    u16*   wih0Bt = (u16*)  alloc(448 * 32 * 2);
    u16*   wih1Bt = (u16*)  alloc((size_t)9  * 448 * 128 * 2);
    u16*   whh1Bt = (u16*)  alloc((size_t)10 * 448 * 128 * 2);
    u16*   wih2Bt = (u16*)  alloc((size_t)10 * 448 * 128 * 2);
    u16*   whh2Bt = (u16*)  alloc((size_t)10 * 448 * 128 * 2);
    u16*   cv1Bt  = (u16*)  alloc((size_t)9 * 112 * 128 * 2);
    u16*   cv2Bt  = (u16*)  alloc((size_t)5 * 112 * 128 * 2);
    u16*   A0     = (u16*)  alloc((size_t)Tlen * 256 * 32 * 2);
    u16*   ysA    = (u16*)  alloc((size_t)Tlen * 256 * 128 * 2);
    u16*   ysB    = (u16*)  alloc((size_t)Tlen * 256 * 128 * 2);
    u16*   Arelu  = (u16*)  alloc((size_t)Tlen * 256 * 128 * 2);
    u16*   G      = (u16*)  alloc((size_t)Tlen * 16 * NT * 256 * 2);
    float* convC  = (float*)alloc((size_t)Tlen * 256 * 112 * 4);

    hipMemsetAsync(zerosH, 0, 256 * 128 * 2, stream);
    hipMemsetAsync(ysA, 0, (size_t)Tlen * 256 * 128 * 2, stream);
    hipMemsetAsync(ysB, 0, (size_t)Tlen * 256 * 128 * 2, stream);

    // ---- pack weights / inputs -------------------------------------------
    pack_x<<<(Tlen * 256 * 32 + 255) / 256, 256, 0, stream>>>(x, A0);
    pack_gate_weight<<<(448 * 32 + 255) / 256, 256, 0, stream>>>(wih0, wih0Bt, 4, 32);
    for (int l = 0; l < 9; l++)
        pack_gate_weight<<<224, 256, 0, stream>>>(wih1 + (size_t)l * 40000,
                                                  wih1Bt + (size_t)l * 57344, 100, 128);
    for (int l = 0; l < 10; l++) {
        pack_gate_weight<<<224, 256, 0, stream>>>(whh1 + (size_t)l * 40000,
                                                  whh1Bt + (size_t)l * 57344, 100, 128);
        pack_gate_weight<<<224, 256, 0, stream>>>(wih2 + (size_t)l * 40000,
                                                  wih2Bt + (size_t)l * 57344, 100, 128);
        pack_gate_weight<<<224, 256, 0, stream>>>(whh2 + (size_t)l * 40000,
                                                  whh2Bt + (size_t)l * 57344, 100, 128);
        pack_bias<<<2, 256, 0, stream>>>(bih1 + l * 400, bhh1 + l * 400, bias1 + l * 448);
        pack_bias<<<2, 256, 0, stream>>>(bih2 + l * 400, bhh2 + l * 400, bias2 + l * 448);
    }
    pack_conv<<<(9 * 112 * 128 + 255) / 256, 256, 0, stream>>>(c1w, cv1Bt, 9);
    pack_conv<<<(5 * 112 * 128 + 255) / 256, 256, 0, stream>>>(c2w, cv2Bt, 5);

    const int Mtiles = Tlen * 16;       // 8192
    const int Mrows  = Tlen * 256;      // 131072
    const int nact   = Tlen * 256 * 128;

    // ---- LSTM stack 1 -----------------------------------------------------
    u16* in = A0; int lda = 32;
    u16* ping = ysA;
    for (int l = 0; l < 10; l++) {
        const u16* wi = (l == 0) ? wih0Bt : (wih1Bt + (size_t)(l - 1) * 57344);
        gemm_tn<<<(Mtiles * NT + 7) / 8, 256, 0, stream>>>(
            in, lda, wi, (l == 0) ? 32 : 128, Mtiles, NT, (l == 0) ? 1 : 4,
            Mrows, 0, nullptr, 0, 0, G);
        lstm_scan<<<16, 32, 0, stream>>>(
            G, whh1Bt + (size_t)l * 57344, bias1 + l * 448,
            zerosH, nullptr, 0, ping,
            hfin + (size_t)l * 32768, cbuf + (size_t)l * 28672, Tlen);
        in = ping; lda = 128;
        ping = (in == ysA) ? ysB : ysA;
    }

    // ---- relu + conv1 (9 shifted GEMM-accumulates) ------------------------
    relu_bf16<<<(nact + 255) / 256, 256, 0, stream>>>(in, Arelu, nact);
    for (int tap = 0; tap < 9; tap++)
        gemm_tn<<<(Mtiles * 7 + 7) / 8, 256, 0, stream>>>(
            Arelu, 128, cv1Bt + (size_t)tap * 14336, 128, Mtiles, 7, 4,
            Mrows, (tap - 4) * 256, convC, 112, (tap != 0) ? 1 : 0, nullptr);
    conv_finish<<<(nact + 255) / 256, 256, 0, stream>>>(convC, c1b, ping, Mrows);

    // ---- LSTM stack 2 (init from stack-1 finals) --------------------------
    in = ping; ping = (in == ysA) ? ysB : ysA;
    for (int l = 0; l < 10; l++) {
        gemm_tn<<<(Mtiles * NT + 7) / 8, 256, 0, stream>>>(
            in, 128, wih2Bt + (size_t)l * 57344, 128, Mtiles, NT, 4,
            Mrows, 0, nullptr, 0, 0, G);
        lstm_scan<<<16, 32, 0, stream>>>(
            G, whh2Bt + (size_t)l * 57344, bias2 + l * 448,
            hfin + (size_t)l * 32768, cbuf + (size_t)l * 28672, 1, ping,
            hfin + (size_t)l * 32768, cbuf + (size_t)l * 28672, Tlen);
        in = ping;
        ping = (in == ysA) ? ysB : ysA;
    }

    // ---- relu + conv2 + linear head ---------------------------------------
    relu_bf16<<<(nact + 255) / 256, 256, 0, stream>>>(in, Arelu, nact);
    for (int tap = 0; tap < 5; tap++)
        gemm_tn<<<(Mtiles * 7 + 7) / 8, 256, 0, stream>>>(
            Arelu, 128, cv2Bt + (size_t)tap * 14336, 128, Mtiles, 7, 4,
            Mrows, (tap - 2) * 256, convC, 112, (tap != 0) ? 1 : 0, nullptr);
    linear_out<<<(Tlen * 256 * 4 + 255) / 256, 256, 0, stream>>>(convC, c2b, lw, lb, out);
}
